// GATModel_66408784331011
// MI455X (gfx1250) — matrix operand order repaired and checked by reading, not compiled, not run on hardware
//
#include <hip/hip_runtime.h>
#include <hip/hip_bf16.h>

typedef __attribute__((ext_vector_type(16))) _Float16 v16h;
typedef __attribute__((ext_vector_type(8)))  float    v8f;

union AF { v16h v; uint4 q[2]; };

__device__ __forceinline__ unsigned ordf(float f) {
  unsigned u = __float_as_uint(f);
  return (u & 0x80000000u) ? ~u : (u | 0x80000000u);
}
__device__ __forceinline__ float unordf(unsigned u) {
  return __uint_as_float((u & 0x80000000u) ? (u & 0x7FFFFFFFu) : ~u);
}

// ---------------------------------------------------------------------------
// Pack f32 weight [K,N] (row-major) into WMMA B fragments (f16), tile (kt,nt):
// lane n = nt*16 + (lane&15), klo = kt*32 + (lane>>4)*8, vgpr v -> k=klo+2(v&3)+16(v>>2).
// ---------------------------------------------------------------------------
__global__ void k_pack_w(const float* __restrict__ W, _Float16* __restrict__ out,
                         int K, int Ncol) {
  int tid = blockIdx.x * blockDim.x + threadIdx.x;
  int NT = Ncol >> 4, KT = K >> 5;
  int total = KT * NT * 32;
  if (tid >= total) return;
  int lane = tid & 31, t = tid >> 5;
  int kt = t / NT, nt = t - kt * NT;
  int k0 = kt * 32 + ((lane >> 4) << 3);
  int n  = (nt << 4) + (lane & 15);
  _Float16* o = out + ((size_t)t * 32 + lane) * 16;
#pragma unroll
  for (int v = 0; v < 8; ++v) {
    int k = k0 + 2 * (v & 3) + ((v >> 2) << 4);
    o[2 * v]     = (_Float16)W[(size_t)k * Ncol + n];
    o[2 * v + 1] = (_Float16)W[(size_t)(k + 1) * Ncol + n];
  }
}

// proj[k,h] = sum_c W[k, h*64+c] * att[h*64+c]   (W row length 256, H=4, C=64)
__global__ void k_proj(const float* __restrict__ W, const float* __restrict__ att,
                       float* __restrict__ proj, int K) {
  int tid = blockIdx.x * blockDim.x + threadIdx.x;
  if (tid >= K * 4) return;
  int k = tid >> 2, h = tid & 3;
  float s = 0.f;
  for (int c = 0; c < 64; ++c) s += W[(size_t)k * 256 + h * 64 + c] * att[h * 64 + c];
  proj[k * 4 + h] = s;
}

__global__ void k_f2h(const float* __restrict__ in, _Float16* __restrict__ out, int n) {
  int i = blockIdx.x * blockDim.x + threadIdx.x;
  if (i < n) out[i] = (_Float16)in[i];
}

// ---------------------------------------------------------------------------
// Generic WMMA GEMM: C[M,Ncol] = act(A[M,K] * B + bias). A in f16 row-major,
// B pre-packed fragments. One wave -> 16x64 output tile (4 D fragments).
// ---------------------------------------------------------------------------
__global__ void k_gemm(const _Float16* __restrict__ A, const _Float16* __restrict__ Bp,
                       const float* __restrict__ bias, float* __restrict__ Cf,
                       _Float16* __restrict__ Ch, int M, int K, int Ncol, int relu) {
  int lane = threadIdx.x & 31;
  int wid = (blockIdx.x * blockDim.x + threadIdx.x) >> 5;
  int nb = Ncol >> 6;
  int tiles = (M >> 4) * nb;
  if (wid >= tiles) return;
  int mt = wid / nb, ng = wid - mt * nb;
  int m0 = mt << 4;
  int NT = Ncol >> 4, KT = K >> 5;
  const _Float16* arow = A + (size_t)(m0 + (lane & 15)) * K + ((lane >> 4) << 3);
  v8f acc[4] = {};
  for (int kt = 0; kt < KT; ++kt) {
    AF a;
    a.q[0] = *(const uint4*)(arow + kt * 32);
    a.q[1] = *(const uint4*)(arow + kt * 32 + 16);
#pragma unroll
    for (int j = 0; j < 4; ++j) {
      const _Float16* bp = Bp + (((size_t)kt * NT + (ng << 2) + j) * 32 + lane) * 16;
      AF b;
      b.q[0] = *(const uint4*)bp;
      b.q[1] = *(const uint4*)(bp + 8);
      acc[j] = __builtin_amdgcn_wmma_f32_16x16x32_f16(false, a.v, false, b.v,
                                                      (short)0, acc[j], false, false);
    }
  }
  int rbase = m0 + ((lane >> 4) << 3);
  int cbase = (ng << 6) + (lane & 15);
#pragma unroll
  for (int j = 0; j < 4; ++j) {
    int col = cbase + j * 16;
    float b = bias ? bias[col] : 0.f;
#pragma unroll
    for (int v = 0; v < 8; ++v) {
      float val = acc[j][v] + b;
      if (relu) val = fmaxf(val, 0.f);
      size_t idx = (size_t)(rbase + v) * Ncol + col;
      if (Cf) Cf[idx] = val;
      if (Ch) Ch[idx] = (_Float16)val;
    }
  }
}

// ---------------------------------------------------------------------------
// Fused edge encoder: eh = relu(ea @ We + be); accumulate self-loop mean,
// emit a_edg for both layers. eh is never stored globally.
// ---------------------------------------------------------------------------
__global__ void k_edge_enc(const float* __restrict__ EA, const float* __restrict__ We,
                           const float* __restrict__ be, const int* __restrict__ dst,
                           const float* __restrict__ pe0, const float* __restrict__ pe1,
                           float* __restrict__ loop_sum, float* __restrict__ cnt,
                           float* __restrict__ ae0, float* __restrict__ ae1, int E) {
  __shared__ float ea[4][64];
  __shared__ float ehs[4][64];
  int g = threadIdx.x >> 6, c = threadIdx.x & 63;
  int e = blockIdx.x * 4 + g;
  bool ok = (e < E);
  if (ok) ea[g][c] = EA[(size_t)e * 64 + c];
  __syncthreads();
  float acc = be[c];
  for (int k = 0; k < 64; ++k) acc += ea[g][k] * We[k * 64 + c];
  float eh = fmaxf(acc, 0.f);
  ehs[g][c] = eh;
  if (ok) {
    int d = dst[e];
    atomicAdd(&loop_sum[(size_t)d * 64 + c], eh);
    if (c == 0) atomicAdd(&cnt[d], 1.f);
  }
  __syncthreads();
  if (ok && c < 8) {
    int h = c & 3;
    const float* pe = (c < 4) ? pe0 : pe1;
    float s = 0.f;
    for (int k = 0; k < 64; ++k) s += ehs[g][k] * pe[k * 4 + h];
    ((c < 4) ? ae0 : ae1)[(size_t)e * 4 + h] = s;
  }
}

__global__ void k_loop_div(float* __restrict__ ls, const float* __restrict__ cnt, int N) {
  int gid = blockIdx.x * blockDim.x + threadIdx.x;
  if (gid >= N * 64) return;
  ls[gid] = ls[gid] / fmaxf(cnt[gid >> 6], 1.f);
}

__global__ void k_loop_aedg(const float* __restrict__ la, const float* __restrict__ pe0,
                            const float* __restrict__ pe1, float* __restrict__ o0,
                            float* __restrict__ o1, int N) {
  int gid = blockIdx.x * blockDim.x + threadIdx.x;
  if (gid >= N * 8) return;
  int n = gid >> 3, j = gid & 7, h = j & 3;
  const float* pe = (j < 4) ? pe0 : pe1;
  float s = 0.f;
  for (int k = 0; k < 64; ++k) s += la[(size_t)n * 64 + k] * pe[k * 4 + h];
  ((j < 4) ? o0 : o1)[(size_t)n * 4 + h] = s;
}

// a_src/a_dst = xh @ proj  (one wave per node, wave32 shuffle reduction)
__global__ void k_adot(const _Float16* __restrict__ X, const float* __restrict__ ps,
                       const float* __restrict__ pd, float* __restrict__ as_,
                       float* __restrict__ ad_, int N, int in) {
  int t = blockIdx.x * blockDim.x + threadIdx.x;
  int wid = t >> 5, lane = t & 31;
  if (wid >= N) return;
  float s[4] = {0, 0, 0, 0}, d[4] = {0, 0, 0, 0};
  const _Float16* row = X + (size_t)wid * in;
  for (int k = lane; k < in; k += 32) {
    float xv = (float)row[k];
#pragma unroll
    for (int h = 0; h < 4; ++h) { s[h] += xv * ps[k * 4 + h]; d[h] += xv * pd[k * 4 + h]; }
  }
#pragma unroll
  for (int h = 0; h < 4; ++h)
    for (int off = 16; off > 0; off >>= 1) {
      s[h] += __shfl_down(s[h], off, 32);
      d[h] += __shfl_down(d[h], off, 32);
    }
  if (lane == 0) {
#pragma unroll
    for (int h = 0; h < 4; ++h) { as_[wid * 4 + h] = s[h]; ad_[wid * 4 + h] = d[h]; }
  }
}

__global__ void k_alpha_max(const int* __restrict__ src, const int* __restrict__ dst,
                            const float* __restrict__ as_, const float* __restrict__ ad_,
                            const float* __restrict__ ae, const float* __restrict__ ael,
                            float* __restrict__ ab, unsigned* __restrict__ mx,
                            int E, int N) {
  int gid = blockIdx.x * blockDim.x + threadIdx.x;
  if (gid >= (E + N) * 4) return;
  int i = gid >> 2, h = gid & 3;
  int s, d; float aedge;
  if (i < E) { s = src[i]; d = dst[i]; aedge = ae[(size_t)i * 4 + h]; }
  else { s = d = i - E; aedge = ael[(size_t)(i - E) * 4 + h]; }
  float a = as_[s * 4 + h] + ad_[d * 4 + h] + aedge;
  a = (a > 0.f) ? a : 0.2f * a;  // leaky_relu(0.2)
  ab[(size_t)i * 4 + h] = a;
  atomicMax(&mx[d * 4 + h], ordf(a));
}

__global__ void k_alpha_exp(const int* __restrict__ dst, float* __restrict__ ab,
                            const unsigned* __restrict__ mx, float* __restrict__ ssum,
                            int E, int N) {
  int gid = blockIdx.x * blockDim.x + threadIdx.x;
  if (gid >= (E + N) * 4) return;
  int i = gid >> 2, h = gid & 3;
  int d = (i < E) ? dst[i] : (i - E);
  float ex = __expf(ab[(size_t)i * 4 + h] - unordf(mx[d * 4 + h]));
  ab[(size_t)i * 4 + h] = ex;
  atomicAdd(&ssum[d * 4 + h], ex);
}

// out[dst] += alpha * h[src]   (one block per edge, 256 channels)
__global__ void k_agg(const int* __restrict__ src, const int* __restrict__ dst,
                      const float* __restrict__ ab, const float* __restrict__ ssum,
                      const float* __restrict__ H, float* __restrict__ agg, int E) {
  int i = blockIdx.x, c = threadIdx.x;
  int s, d;
  if (i < E) { s = src[i]; d = dst[i]; } else { s = d = i - E; }
  int h = c >> 6;
  float w = ab[(size_t)i * 4 + h] / ssum[d * 4 + h];
  atomicAdd(&agg[(size_t)d * 256 + c], H[(size_t)s * 256 + c] * w);
}

__global__ void k_final(const float* __restrict__ agg, const float* __restrict__ bias,
                        float* __restrict__ Cf, _Float16* __restrict__ Ch, int N) {
  int gid = blockIdx.x * blockDim.x + threadIdx.x;
  if (gid >= N * 256) return;
  float v = fmaxf(agg[gid] + bias[gid & 255], 0.f);
  if (Cf) Cf[gid] = v;
  Ch[gid] = (_Float16)v;
}

__global__ void k_zero1(float* p) { if (threadIdx.x == 0 && blockIdx.x == 0) *p = 0.f; }

// ---------------------------------------------------------------------------
// Edge head: hmid = relu([xh[src],xh[dst]] @ W1 + b1) via WMMA (K=512 gathered),
// fused score = sigmoid(hmid @ w2 + b2) through LDS, mean via atomicAdd.
// ---------------------------------------------------------------------------
__global__ void k_edge_head(const int* __restrict__ src, const int* __restrict__ dst,
                            const _Float16* __restrict__ X, const _Float16* __restrict__ Bp,
                            const float* __restrict__ b1, const float* __restrict__ w2,
                            const float* __restrict__ b2, float* __restrict__ scores,
                            float* __restrict__ psum, int E, float invE) {
  __shared__ float hm[8][16][64];
  int lane = threadIdx.x & 31, w = threadIdx.x >> 5;
  int wid = blockIdx.x * 8 + w;
  int tiles = E >> 4;
  bool active = wid < tiles;
  v8f acc[4] = {};
  if (active) {
    int e = (wid << 4) + (lane & 15);
    int s = src[e], d = dst[e];
    for (int kt = 0; kt < 16; ++kt) {
      int ks0 = kt * 32 + ((lane >> 4) << 3);
      int ks1 = ks0 + 16;
      const _Float16* r0 = X + (size_t)(ks0 < 256 ? s : d) * 256 + (ks0 & 255);
      const _Float16* r1 = X + (size_t)(ks1 < 256 ? s : d) * 256 + (ks1 & 255);
      AF a;
      a.q[0] = *(const uint4*)r0;
      a.q[1] = *(const uint4*)r1;
#pragma unroll
      for (int j = 0; j < 4; ++j) {
        const _Float16* bp = Bp + (((size_t)kt * 4 + j) * 32 + lane) * 16;
        AF b;
        b.q[0] = *(const uint4*)bp;
        b.q[1] = *(const uint4*)(bp + 8);
        acc[j] = __builtin_amdgcn_wmma_f32_16x16x32_f16(false, a.v, false, b.v,
                                                        (short)0, acc[j], false, false);
      }
    }
    int row0 = (lane >> 4) << 3;
#pragma unroll
    for (int j = 0; j < 4; ++j) {
      int col = j * 16 + (lane & 15);
      float b = b1[col];
#pragma unroll
      for (int v = 0; v < 8; ++v) hm[w][row0 + v][col] = fmaxf(acc[j][v] + b, 0.f);
    }
  }
  __syncthreads();
  int t = threadIdx.x;
  if (t < 128) {
    int ww = t >> 4, m = t & 15;
    int tile = blockIdx.x * 8 + ww;
    if (tile < tiles) {
      float sacc = b2[0];
      for (int j = 0; j < 64; ++j) sacc += hm[ww][m][j] * w2[j];
      float sc = 1.f / (1.f + __expf(-sacc));
      scores[tile * 16 + m] = sc;
      atomicAdd(psum, sc * invE);
    }
  }
}

// ---------------------------------------------------------------------------
extern "C" void kernel_launch(void* const* d_in, const int* in_sizes, int n_in,
                              void* d_out, int out_size, void* d_ws, size_t ws_size,
                              hipStream_t stream) {
  (void)n_in; (void)out_size; (void)ws_size;
  const float* x      = (const float*)d_in[0];
  const float* eattr  = (const float*)d_in[1];
  const int*   ei     = (const int*)d_in[2];
  const float* enc_wn = (const float*)d_in[3];
  const float* enc_bn = (const float*)d_in[4];
  const float* enc_we = (const float*)d_in[5];
  const float* enc_be = (const float*)d_in[6];
  const float* w0     = (const float*)d_in[7];
  const float* att_s0 = (const float*)d_in[9];
  const float* att_d0 = (const float*)d_in[10];
  const float* att_e0 = (const float*)d_in[11];
  const float* bias0  = (const float*)d_in[12];
  const float* w1     = (const float*)d_in[13];
  const float* wedge0 = (const float*)d_in[8];
  const float* wedge1 = (const float*)d_in[14];
  const float* att_s1 = (const float*)d_in[15];
  const float* att_d1 = (const float*)d_in[16];
  const float* att_e1 = (const float*)d_in[17];
  const float* bias1  = (const float*)d_in[18];
  const float* eh_w1  = (const float*)d_in[19];
  const float* eh_b1  = (const float*)d_in[20];
  const float* eh_w2  = (const float*)d_in[21];
  const float* eh_b2  = (const float*)d_in[22];

  const int N = in_sizes[0] / 128;
  const int E = in_sizes[1] / 64;
  const int TE = E + N;
  const int* src = ei;
  const int* dst = ei + E;

  float* out_scores = (float*)d_out;      // [E]
  float* out_path   = out_scores + E;     // [1]
  float* out_xh     = out_path + 1;       // [N*256]

  size_t off = 0;
  auto alloc = [&](size_t bytes) -> void* {
    void* p = (char*)d_ws + off;
    off += (bytes + 255) & ~(size_t)255;
    return p;
  };
  _Float16* bp_wn = (_Float16*)alloc((size_t)128 * 64 * 2);
  _Float16* bp_w0 = (_Float16*)alloc((size_t)64 * 256 * 2);
  _Float16* bp_w1 = (_Float16*)alloc((size_t)256 * 256 * 2);
  _Float16* bp_eh = (_Float16*)alloc((size_t)512 * 64 * 2);
  float* ps0 = (float*)alloc(64 * 4 * 4);
  float* pd0 = (float*)alloc(64 * 4 * 4);
  float* pe0 = (float*)alloc(64 * 4 * 4);
  float* ps1 = (float*)alloc(256 * 4 * 4);
  float* pd1 = (float*)alloc(256 * 4 * 4);
  float* pe1 = (float*)alloc(64 * 4 * 4);
  _Float16* xh_   = (_Float16*)alloc((size_t)N * 128 * 2);
  _Float16* x0h   = (_Float16*)alloc((size_t)N * 64 * 2);
  float* hbuf     = (float*)alloc((size_t)N * 256 * 4);
  float* aggbuf   = (float*)alloc((size_t)N * 256 * 4);
  _Float16* x1h   = (_Float16*)alloc((size_t)N * 256 * 2);
  _Float16* x2h   = (_Float16*)alloc((size_t)N * 256 * 2);
  float* loop_sum = (float*)alloc((size_t)N * 64 * 4);
  float* cnt      = (float*)alloc((size_t)N * 4);
  float* ae0      = (float*)alloc((size_t)E * 4 * 4);
  float* ae1      = (float*)alloc((size_t)E * 4 * 4);
  float* ael0     = (float*)alloc((size_t)N * 4 * 4);
  float* ael1     = (float*)alloc((size_t)N * 4 * 4);
  float* asrc     = (float*)alloc((size_t)N * 4 * 4);
  float* adst     = (float*)alloc((size_t)N * 4 * 4);
  float* abuf     = (float*)alloc((size_t)TE * 4 * 4);
  unsigned* mx    = (unsigned*)alloc((size_t)N * 4 * 4);
  float* ssum     = (float*)alloc((size_t)N * 4 * 4);

  auto cdiv = [](int a, int b) { return (a + b - 1) / b; };
  auto packT = [&](int K, int Nc) { return (K / 32) * (Nc / 16) * 32; };

  // Weight packing + attention projections
  k_pack_w<<<cdiv(packT(128, 64), 256), 256, 0, stream>>>(enc_wn, bp_wn, 128, 64);
  k_pack_w<<<cdiv(packT(64, 256), 256), 256, 0, stream>>>(w0, bp_w0, 64, 256);
  k_pack_w<<<cdiv(packT(256, 256), 256), 256, 0, stream>>>(w1, bp_w1, 256, 256);
  k_pack_w<<<cdiv(packT(512, 64), 256), 256, 0, stream>>>(eh_w1, bp_eh, 512, 64);
  k_proj<<<cdiv(64 * 4, 256), 256, 0, stream>>>(w0, att_s0, ps0, 64);
  k_proj<<<cdiv(64 * 4, 256), 256, 0, stream>>>(w0, att_d0, pd0, 64);
  k_proj<<<cdiv(64 * 4, 256), 256, 0, stream>>>(wedge0, att_e0, pe0, 64);
  k_proj<<<cdiv(256 * 4, 256), 256, 0, stream>>>(w1, att_s1, ps1, 256);
  k_proj<<<cdiv(256 * 4, 256), 256, 0, stream>>>(w1, att_d1, pd1, 256);
  k_proj<<<cdiv(64 * 4, 256), 256, 0, stream>>>(wedge1, att_e1, pe1, 64);

  hipMemsetAsync(loop_sum, 0, (size_t)N * 64 * 4, stream);
  hipMemsetAsync(cnt, 0, (size_t)N * 4, stream);

  // Node encoder: xh0 = relu(x @ enc_wn + enc_bn)
  k_f2h<<<cdiv(N * 128, 256), 256, 0, stream>>>(x, xh_, N * 128);
  k_gemm<<<cdiv((N / 16) * 1 * 32, 256), 256, 0, stream>>>(
      xh_, bp_wn, enc_bn, nullptr, x0h, N, 128, 64, 1);

  // Fused edge encoder (+ self-loop mean accumulation + a_edg for both layers)
  k_edge_enc<<<cdiv(E, 4), 256, 0, stream>>>(eattr, enc_we, enc_be, dst, pe0, pe1,
                                             loop_sum, cnt, ae0, ae1, E);
  k_loop_div<<<cdiv(N * 64, 256), 256, 0, stream>>>(loop_sum, cnt, N);
  k_loop_aedg<<<cdiv(N * 8, 256), 256, 0, stream>>>(loop_sum, pe0, pe1, ael0, ael1, N);

  // ---- GAT layer 0 ----
  k_gemm<<<cdiv((N / 16) * 4 * 32, 256), 256, 0, stream>>>(
      x0h, bp_w0, nullptr, hbuf, nullptr, N, 64, 256, 0);
  k_adot<<<cdiv(N * 32, 256), 256, 0, stream>>>(x0h, ps0, pd0, asrc, adst, N, 64);
  hipMemsetAsync(mx, 0, (size_t)N * 16, stream);
  hipMemsetAsync(ssum, 0, (size_t)N * 16, stream);
  hipMemsetAsync(aggbuf, 0, (size_t)N * 256 * 4, stream);
  k_alpha_max<<<cdiv(TE * 4, 256), 256, 0, stream>>>(src, dst, asrc, adst, ae0, ael0,
                                                     abuf, mx, E, N);
  k_alpha_exp<<<cdiv(TE * 4, 256), 256, 0, stream>>>(dst, abuf, mx, ssum, E, N);
  k_agg<<<TE, 256, 0, stream>>>(src, dst, abuf, ssum, hbuf, aggbuf, E);
  k_final<<<cdiv(N * 256, 256), 256, 0, stream>>>(aggbuf, bias0, nullptr, x1h, N);

  // ---- GAT layer 1 ----
  k_gemm<<<cdiv((N / 16) * 4 * 32, 256), 256, 0, stream>>>(
      x1h, bp_w1, nullptr, hbuf, nullptr, N, 256, 256, 0);
  k_adot<<<cdiv(N * 32, 256), 256, 0, stream>>>(x1h, ps1, pd1, asrc, adst, N, 256);
  hipMemsetAsync(mx, 0, (size_t)N * 16, stream);
  hipMemsetAsync(ssum, 0, (size_t)N * 16, stream);
  hipMemsetAsync(aggbuf, 0, (size_t)N * 256 * 4, stream);
  k_alpha_max<<<cdiv(TE * 4, 256), 256, 0, stream>>>(src, dst, asrc, adst, ae1, ael1,
                                                     abuf, mx, E, N);
  k_alpha_exp<<<cdiv(TE * 4, 256), 256, 0, stream>>>(dst, abuf, mx, ssum, E, N);
  k_agg<<<TE, 256, 0, stream>>>(src, dst, abuf, ssum, hbuf, aggbuf, E);
  k_final<<<cdiv(N * 256, 256), 256, 0, stream>>>(aggbuf, bias1, out_xh, x2h, N);

  // ---- Edge head ----
  k_zero1<<<1, 1, 0, stream>>>(out_path);
  k_edge_head<<<cdiv(E / 16, 8), 256, 0, stream>>>(src, dst, x2h, bp_eh, eh_b1, eh_w2,
                                                   eh_b2, out_scores, out_path, E,
                                                   1.0f / (float)E);
}
